// RigPoseTransformer_22823456211289
// MI455X (gfx1250) — matrix operand order repaired and checked by reading, not compiled, not run on hardware
//
#include <hip/hip_runtime.h>
#include <cmath>

// ---- problem constants (match reference) ----
#define BB   4
#define NT_  2048
#define NA_  2048
#define NP_  8192      // B*NT == B*NA (flattened point count)
#define CC   128
#define HH   8
#define KQQ  16
#define KQC  64

typedef __attribute__((ext_vector_type(16))) __bf16 v16bf;
typedef __attribute__((ext_vector_type(8)))  __bf16 v8bf;
typedef __attribute__((ext_vector_type(8)))  float  v8f;

static __device__ __forceinline__ __bf16 f2bf(float f) {
  union { float f; unsigned u; } c; c.f = f;
  unsigned r = (c.u + 0x7FFFu + ((c.u >> 16) & 1u)) >> 16;
  union { unsigned short s; __bf16 b; } o; o.s = (unsigned short)r;
  return o.b;
}

static __device__ __forceinline__ v8bf ld8(const __bf16* p) {
  return *(const v8bf*)p;   // 16B vector load (global_load_b128 / ds_load_b128)
}
static __device__ __forceinline__ v16bf concat8(v8bf lo, v8bf hi) {
  return __builtin_shufflevector(lo, hi, 0,1,2,3,4,5,6,7,8,9,10,11,12,13,14,15);
}

// A fragment (16x32 bf16, row-major source): lane holds rows of 8+8 consecutive K
static __device__ __forceinline__ v16bf load_afrag(const __bf16* rowp, int lane) {
  const int base = ((lane >> 4) & 1) << 3;        // 0 or 8
  return concat8(ld8(rowp + base), ld8(rowp + base + 16));
}
// B fragment (32x16 bf16) from K-contiguous source (transposed weights / fb rows)
static __device__ __forceinline__ v16bf load_bfrag(const __bf16* colp, int lane) {
  const int base = ((lane >> 4) & 1) << 3;        // 0 or 8
  return concat8(ld8(colp + base), ld8(colp + base + 16));
}

static __device__ __forceinline__ v8f wmma_bf16(v16bf a, v16bf b, v8f c) {
  return __builtin_amdgcn_wmma_f32_16x16x32_bf16(false, a, false, b, (short)0, c, false, false);
}

// ================= conversion / weight-prep kernels ======================================
__global__ __launch_bounds__(256) void cvt_bf16_kernel(
    const float* __restrict__ in, __bf16* __restrict__ out, long n)
{
  const long i = (long)blockIdx.x * 256 + threadIdx.x;
  if (i < n) out[i] = f2bf(in[i]);
}

// Wt[n*128 + k] = bf16(W[k*Nreal + n]); rows n >= Nreal are zero padding
__global__ __launch_bounds__(256) void transpose_w_kernel(
    const float* __restrict__ W, __bf16* __restrict__ Wt, int Npad, int Nreal)
{
  const long i = (long)blockIdx.x * 256 + threadIdx.x;
  if (i >= (long)Npad * 128) return;
  const int n = (int)(i >> 7), k = (int)(i & 127);
  Wt[i] = (n < Nreal) ? f2bf(W[(long)k * Nreal + n]) : f2bf(0.f);
}

// ======= GEMM: Y/Ybf = [resid +] A_bf(Mx128) @ Wt_bf(N x 128, pre-transposed) [+ bias] ===
__global__ __launch_bounds__(256) void gemm_kernel(
    const __bf16* __restrict__ A, const __bf16* __restrict__ Wt,
    const float* __restrict__ bias, const float* __restrict__ resid,
    float* __restrict__ Y, __bf16* __restrict__ Ybf, int M, int N)
{
  const int lane = threadIdx.x & 31;
  const int wave = threadIdx.x >> 5;
  const int tile = blockIdx.x * (blockDim.x >> 5) + wave;
  const int tiles_n = N >> 4;
  const int total = (M >> 4) * tiles_n;
  if (tile >= total) return;
  const int tm = tile / tiles_n, tn = tile % tiles_n;
  const int arow = (tm << 4) + (lane & 15);
  const int bcol = (tn << 4) + (lane & 15);
  const __bf16* pa = A + (long)arow * CC;
  const __bf16* pb = Wt + (long)bcol * CC;
  __builtin_prefetch(pa + 64, 0, 1);
  v8f acc = {};
#pragma unroll
  for (int kb = 0; kb < CC; kb += 32) {
    const v16bf av = load_afrag(pa + kb, lane);
    const v16bf bv = load_bfrag(pb + kb, lane);
    acc = wmma_bf16(av, bv, acc);
  }
  const float bs = bias ? bias[bcol] : 0.f;
#pragma unroll
  for (int r = 0; r < 8; ++r) {
    const int m = (tm << 4) + r + ((lane >> 4) << 3);
    float v = acc[r] + bs;
    if (resid) v += resid[(long)m * N + bcol];
    if (Y)   Y[(long)m * N + bcol] = v;
    if (Ybf) Ybf[(long)m * N + bcol] = f2bf(v);
  }
}

// ========== similarity: sim[b,n,m] = scale * fa[b,n,:].fb[b,m,:]  (both bf16) ============
__global__ __launch_bounds__(256) void sim_kernel(
    const __bf16* __restrict__ fa, const __bf16* __restrict__ fb,
    float* __restrict__ sim, float scale)
{
  const int lane = threadIdx.x & 31;
  const int wave = threadIdx.x >> 5;
  const long tile = (long)blockIdx.x * (blockDim.x >> 5) + wave;
  const int tn_per = NA_ >> 4;                 // 128
  const int tiles_per_b = (NT_ >> 4) * tn_per; // 16384
  if (tile >= (long)BB * tiles_per_b) return;
  const int b = (int)(tile / tiles_per_b);
  const int t = (int)(tile % tiles_per_b);
  const int tm = t / tn_per, tn = t % tn_per;
  const __bf16* fab = fa + (long)b * NT_ * CC;
  const __bf16* fbb = fb + (long)b * NA_ * CC;
  float* simb = sim + (long)b * NT_ * NA_;
  const int rq = (tm << 4) + (lane & 15);
  const int ra = (tn << 4) + (lane & 15);
  const __bf16* pa = fab + (long)rq * CC;
  const __bf16* pb = fbb + (long)ra * CC;   // fb row is K-contiguous => B fragment directly
  __builtin_prefetch(pa + 64, 0, 1);
  v8f acc = {};
#pragma unroll
  for (int kb = 0; kb < CC; kb += 32) {
    const v16bf av = load_afrag(pa + kb, lane);
    const v16bf bv = load_bfrag(pb + kb, lane);
    acc = wmma_bf16(av, bv, acc);
  }
#pragma unroll
  for (int r = 0; r < 8; ++r) {
    const int n = (tm << 4) + r + ((lane >> 4) << 3);
    const int m = (tn << 4) + (lane & 15);
    simb[(long)n * NA_ + m] = acc[r] * scale;
  }
}

// ======================= brute-force KNN (one workgroup per query) =======================
template <int K>
__global__ __launch_bounds__(256) void knn_kernel(
    const float* __restrict__ qc, const float* __restrict__ bc,
    int nq_per_b, int nb_per_b, int* __restrict__ out)
{
  __shared__ float dist[2048];
  __shared__ float rv[8];
  __shared__ int   ri[8];
  __shared__ float qx[3];
  const int q = blockIdx.x;
  const int b = q / nq_per_b;
  const int base0 = b * nb_per_b;
  const int tid = threadIdx.x;
  const int lane = tid & 31, wave = tid >> 5;
  if (tid < 3) qx[tid] = qc[(long)q * 3 + tid];
  __syncthreads();
  const float q0 = qx[0], q1 = qx[1], q2 = qx[2];
  for (int j = tid; j < nb_per_b; j += 256) {
    const float dx = q0 - bc[(long)(base0 + j) * 3 + 0];
    const float dy = q1 - bc[(long)(base0 + j) * 3 + 1];
    const float dz = q2 - bc[(long)(base0 + j) * 3 + 2];
    dist[j] = dx * dx + dy * dy + dz * dz;
  }
  __syncthreads();
  for (int s = 0; s < K; ++s) {
    float bv = 3.4e38f; int bi = 0;
    for (int j = tid; j < nb_per_b; j += 256) {
      const float d = dist[j];
      if (d < bv) { bv = d; bi = j; }
    }
    for (int off = 16; off > 0; off >>= 1) {
      const float ov = __shfl_down(bv, off, 32);
      const int   oi = __shfl_down(bi, off, 32);
      if (ov < bv) { bv = ov; bi = oi; }
    }
    if (lane == 0) { rv[wave] = bv; ri[wave] = bi; }
    __syncthreads();
    if (tid == 0) {
      float fv = rv[0]; int fi = ri[0];
      for (int w = 1; w < 8; ++w) if (rv[w] < fv) { fv = rv[w]; fi = ri[w]; }
      out[(long)q * K + s] = base0 + fi;
      dist[fi] = 3.4e38f;
    }
    __syncthreads();
  }
}

// ======= in-LDS GEMM: out = [relu](A_lds(Kx128,bf16) @ Wt_bf(128x128 transposed) + bias) =
template <int K, bool RELU, bool TOBF>
static __device__ __forceinline__ void lds_gemm128(
    const __bf16* A, const __bf16* __restrict__ Wt, const float* __restrict__ bias,
    float* outf, __bf16* outb, int lane, int wave)
{
  constexpr int TPW = (K / 16) * 8 / 8; // tiles per wave (8 waves)
  v8f accs[TPW];
  for (int t = 0; t < TPW; ++t) {
    const int tile = wave * TPW + t;
    const int tm = tile >> 3, tn = tile & 7;
    const __bf16* pa = A + ((tm << 4) + (lane & 15)) * 128;
    const __bf16* pb = Wt + ((tn << 4) + (lane & 15)) * 128;
    v8f acc = {};
#pragma unroll
    for (int kb = 0; kb < 128; kb += 32) {
      const v16bf av = load_afrag(pa + kb, lane);   // ds_load_b128 x2
      const v16bf bv = load_bfrag(pb + kb, lane);   // global_load_b128 x2
      acc = wmma_bf16(av, bv, acc);
    }
    accs[t] = acc;
  }
  __syncthreads(); // all reads of A done before any (possibly aliased) writes
  for (int t = 0; t < TPW; ++t) {
    const int tile = wave * TPW + t;
    const int tm = tile >> 3, tn = tile & 7;
    const int bcol = (tn << 4) + (lane & 15);
    const float bs = bias[bcol];
#pragma unroll
    for (int r = 0; r < 8; ++r) {
      const int m = (tm << 4) + r + ((lane >> 4) << 3);
      float v = accs[t][r] + bs;
      if (RELU) v = fmaxf(v, 0.f);
      if (TOBF) outb[m * 128 + bcol] = f2bf(v);
      else      outf[m * 128 + bcol] = v;
    }
  }
  __syncthreads();
}

// ======================= fused PTv2 grouped vector attention =============================
template <int K>
__global__ __launch_bounds__(256) void attn_kernel(
    const float* __restrict__ qproj, const float* __restrict__ kproj,
    const float* __restrict__ vproj, const float* __restrict__ qcoord,
    const float* __restrict__ kvcoord, const int* __restrict__ knn,
    const float* __restrict__ pe_w1, const float* __restrict__ pe_b1,
    const __bf16* __restrict__ pe_w2t, const float* __restrict__ pe_b2,
    const __bf16* __restrict__ w_w1t,  const float* __restrict__ w_b1,
    const __bf16* __restrict__ w_w2t,  const float* __restrict__ w_b2,
    __bf16* __restrict__ attout_bf)
{
  __shared__ __bf16 s_a[K * 128];   // pe1 -> rel -> w1 (A operand, bf16)
  __shared__ float  s_pe[K * 128];  // pe (f32, kept for val = v + pe)
  __shared__ float  s_w[K * HH];
  __shared__ float  s_pos[K * 3];
  __shared__ float  s_q[128];
  __shared__ int    s_idx[K];
  const int n = blockIdx.x;
  const int tid = threadIdx.x;
  const int lane = tid & 31, wave = tid >> 5;

  if (tid < K)   s_idx[tid] = knn[(long)n * K + tid];
  if (tid < 128) s_q[tid] = qproj[(long)n * 128 + tid];
  __syncthreads();
  if (tid < 3 * K) {
    const int k = tid / 3, d = tid % 3;
    s_pos[tid] = qcoord[(long)n * 3 + d] - kvcoord[(long)s_idx[k] * 3 + d];
  }
  __syncthreads();
  // pe1 = relu(pos @ pe_w1 + pe_b1)  (K x 3 @ 3 x 128: VALU)
  for (int e = tid; e < K * 128; e += 256) {
    const int k = e >> 7, c = e & 127;
    float acc = pe_b1[c]
      + s_pos[k * 3 + 0] * pe_w1[c]
      + s_pos[k * 3 + 1] * pe_w1[128 + c]
      + s_pos[k * 3 + 2] * pe_w1[256 + c];
    s_a[e] = f2bf(fmaxf(acc, 0.f));
  }
  __syncthreads();
  // pe = pe1 @ pe_w2 + pe_b2
  lds_gemm128<K, false, false>(s_a, pe_w2t, pe_b2, s_pe, (__bf16*)nullptr, lane, wave);
  // rel = q - k + pe
  for (int e = tid; e < K * 128; e += 256) {
    const int k = e >> 7, c = e & 127;
    const float kk = kproj[(long)s_idx[k] * 128 + c];
    s_a[e] = f2bf(s_q[c] - kk + s_pe[e]);
  }
  __syncthreads();
  // w1 = relu(rel @ w_w1 + w_b1)  (written back into s_a after internal barrier)
  lds_gemm128<K, true, true>(s_a, w_w1t, w_b1, (float*)nullptr, s_a, lane, wave);
  // w = w1 @ w_w2 + w_b2   ([K,128]@[128,8]; w_w2t zero-padded to 16 columns)
  {
    constexpr int TM = K / 16;
    if (wave < TM) {
      const int tm = wave;
      const int ncol = lane & 15;
      const __bf16* pa = s_a + ((tm << 4) + (lane & 15)) * 128;
      const __bf16* pb = w_w2t + ncol * 128;
      v8f acc = {};
#pragma unroll
      for (int kb = 0; kb < 128; kb += 32) {
        const v16bf av = load_afrag(pa + kb, lane);
        const v16bf bv = load_bfrag(pb + kb, lane);
        acc = wmma_bf16(av, bv, acc);
      }
      if (ncol < HH) {
        const float bs = w_b2[ncol];
#pragma unroll
        for (int r = 0; r < 8; ++r) {
          const int m = (tm << 4) + r + ((lane >> 4) << 3);
          s_w[m * HH + ncol] = acc[r] + bs;
        }
      }
    }
  }
  __syncthreads();
  // softmax over neighbors per head
  if (tid < HH) {
    const int h = tid;
    float mx = -3.4e38f;
    for (int k = 0; k < K; ++k) mx = fmaxf(mx, s_w[k * HH + h]);
    float sm = 0.f;
    for (int k = 0; k < K; ++k) { const float e = expf(s_w[k * HH + h] - mx); s_w[k * HH + h] = e; sm += e; }
    const float inv = 1.f / sm;
    for (int k = 0; k < K; ++k) s_w[k * HH + h] *= inv;
  }
  __syncthreads();
  // out[c] = sum_k w[k, c/16] * (v[k,c] + pe[k,c])   -> bf16 (A operand of wo-GEMM)
  if (tid < 128) {
    const int c = tid, h = c >> 4; // C/H = 16
    float acc = 0.f;
    for (int k = 0; k < K; ++k) {
      const float val = vproj[(long)s_idx[k] * 128 + c] + s_pe[k * 128 + c];
      acc += s_w[k * HH + h] * val;
    }
    attout_bf[(long)n * 128 + c] = f2bf(acc);
  }
}

// ======================= dual-softmax statistics & combine ===============================
__global__ __launch_bounds__(256) void row_stats_kernel(
    const float* __restrict__ sim, float* __restrict__ rmax, float* __restrict__ rsum)
{
  __shared__ float red[8];
  __shared__ float bmx;
  const long row = blockIdx.x; // b*NT + n
  const float* s = sim + row * NA_;
  const int tid = threadIdx.x, lane = tid & 31, wave = tid >> 5;
  float mx = -3.4e38f;
  for (int m = tid; m < NA_; m += 256) mx = fmaxf(mx, s[m]);
  for (int off = 16; off > 0; off >>= 1) mx = fmaxf(mx, __shfl_down(mx, off, 32));
  if (lane == 0) red[wave] = mx;
  __syncthreads();
  if (tid == 0) {
    float v = red[0];
    for (int w = 1; w < 8; ++w) v = fmaxf(v, red[w]);
    bmx = v;
  }
  __syncthreads();
  mx = bmx;
  float sm = 0.f;
  for (int m = tid; m < NA_; m += 256) sm += expf(s[m] - mx);
  for (int off = 16; off > 0; off >>= 1) sm += __shfl_down(sm, off, 32);
  __syncthreads();
  if (lane == 0) red[wave] = sm;
  __syncthreads();
  if (tid == 0) {
    float v = 0.f;
    for (int w = 0; w < 8; ++w) v += red[w];
    rmax[row] = mx; rsum[row] = v;
  }
}

__global__ __launch_bounds__(256) void col_stats_kernel(
    const float* __restrict__ sim, float* __restrict__ cmax, float* __restrict__ csum)
{
  const long col = (long)blockIdx.x * 256 + threadIdx.x; // b*NA + m
  if (col >= (long)BB * NA_) return;
  const int b = (int)(col / NA_), m = (int)(col % NA_);
  const float* s = sim + (long)b * NT_ * NA_ + m;
  float mx = -3.4e38f;
  for (int n = 0; n < NT_; ++n) mx = fmaxf(mx, s[(long)n * NA_]);
  float sm = 0.f;
  for (int n = 0; n < NT_; ++n) sm += expf(s[(long)n * NA_] - mx);
  cmax[col] = mx; csum[col] = sm;
}

__global__ __launch_bounds__(256) void conf_kernel(
    const float* __restrict__ sim,
    const float* __restrict__ rmax, const float* __restrict__ rsum,
    const float* __restrict__ cmax, const float* __restrict__ csum,
    float* __restrict__ out)
{
  const long i = (long)blockIdx.x * 256 + threadIdx.x;
  const long nm = (long)NT_ * NA_;
  if (i >= (long)BB * nm) return;
  const int b = (int)(i / nm);
  const long r = i % nm;
  const int n = (int)(r / NA_), m = (int)(r % NA_);
  const float s = sim[i];
  const float p_row = expf(s - rmax[(long)b * NT_ + n]) / rsum[(long)b * NT_ + n]; // softmax axis=2
  const float p_col = expf(s - cmax[(long)b * NA_ + m]) / csum[(long)b * NA_ + m]; // softmax axis=1
  out[i] = p_row * p_col;
}

// ========================================================================================
extern "C" void kernel_launch(void* const* d_in, const int* in_sizes, int n_in,
                              void* d_out, int out_size, void* d_ws, size_t ws_size,
                              hipStream_t stream)
{
  (void)in_sizes; (void)n_in; (void)out_size; (void)ws_size;
  const float* a_coord = (const float*)d_in[0];
  const float* a_feat  = (const float*)d_in[1];
  const float* t_coord = (const float*)d_in[3];
  const float* t_feat  = (const float*)d_in[4];

  // params flatten (jax dict keys sorted): blocks -> [b0:{context,query}, b1:{context,query}], then wm
  // per-set sorted leaves: bo, pe_b1, pe_b2, pe_w1, pe_w2, w_b1, w_b2, w_w1, w_w2, wk, wo, wq, wv
  struct PT { const float *bo,*pe_b1,*pe_b2,*pe_w1,*pe_w2,*w_b1,*w_b2,*w_w1,*w_w2,*wk,*wo,*wq,*wv; };
  PT sets[4];
  for (int s = 0; s < 4; ++s) {
    const int base = 6 + s * 13;
    sets[s].bo    = (const float*)d_in[base + 0];
    sets[s].pe_b1 = (const float*)d_in[base + 1];
    sets[s].pe_b2 = (const float*)d_in[base + 2];
    sets[s].pe_w1 = (const float*)d_in[base + 3];
    sets[s].pe_w2 = (const float*)d_in[base + 4];
    sets[s].w_b1  = (const float*)d_in[base + 5];
    sets[s].w_b2  = (const float*)d_in[base + 6];
    sets[s].w_w1  = (const float*)d_in[base + 7];
    sets[s].w_w2  = (const float*)d_in[base + 8];
    sets[s].wk    = (const float*)d_in[base + 9];
    sets[s].wo    = (const float*)d_in[base + 10];
    sets[s].wq    = (const float*)d_in[base + 11];
    sets[s].wv    = (const float*)d_in[base + 12];
  }
  const float* wm = (const float*)d_in[6 + 4 * 13];

  // ---- workspace carve-out ----
  char* wsb = (char*)d_ws;
  size_t off = 0;
  auto alloc = [&](size_t bytes) -> void* {
    void* p = wsb + off; off += (bytes + 255) & ~(size_t)255; return p;
  };
  const size_t featB  = (size_t)NP_ * CC * sizeof(float);
  const size_t featBh = (size_t)NP_ * CC * sizeof(__bf16);
  const size_t wB     = (size_t)CC * CC * sizeof(__bf16);
  int*   knnqq = (int*)alloc((size_t)NP_ * KQQ * sizeof(int));
  int*   knnqc = (int*)alloc((size_t)NP_ * KQC * sizeof(int));
  int*   knncq = (int*)alloc((size_t)NP_ * KQC * sizeof(int));
  float* t0 = (float*)alloc(featB);
  float* t1 = (float*)alloc(featB);
  float* a0 = (float*)alloc(featB);
  float* a1 = (float*)alloc(featB);
  __bf16* t0b = (__bf16*)alloc(featBh);
  __bf16* t1b = (__bf16*)alloc(featBh);
  __bf16* a0b = (__bf16*)alloc(featBh);
  __bf16* a1b = (__bf16*)alloc(featBh);
  float* qproj  = (float*)alloc(featB);
  float* kproj  = (float*)alloc(featB);
  float* vproj  = (float*)alloc(featB);
  __bf16* attout_bf = (__bf16*)alloc(featBh);
  __bf16* fab = (__bf16*)alloc(featBh);
  __bf16* fbb = (__bf16*)alloc(featBh);
  float* sim  = (float*)alloc((size_t)BB * NT_ * NA_ * sizeof(float));
  float* rmax = (float*)alloc((size_t)BB * NT_ * sizeof(float));
  float* rsum = (float*)alloc((size_t)BB * NT_ * sizeof(float));
  float* cmax = (float*)alloc((size_t)BB * NA_ * sizeof(float));
  float* csum = (float*)alloc((size_t)BB * NA_ * sizeof(float));
  // transposed bf16 weights per set
  struct PTB { __bf16 *wqt,*wkt,*wvt,*wot,*pe_w2t,*w_w1t,*w_w2t; };
  PTB pset[4];
  for (int s = 0; s < 4; ++s) {
    pset[s].wqt    = (__bf16*)alloc(wB);
    pset[s].wkt    = (__bf16*)alloc(wB);
    pset[s].wvt    = (__bf16*)alloc(wB);
    pset[s].wot    = (__bf16*)alloc(wB);
    pset[s].pe_w2t = (__bf16*)alloc(wB);
    pset[s].w_w1t  = (__bf16*)alloc(wB);
    pset[s].w_w2t  = (__bf16*)alloc((size_t)16 * CC * sizeof(__bf16));
  }
  __bf16* wmt = (__bf16*)alloc(wB);

  // ---- weight prep (fp32 -> transposed bf16) ----
  auto tw = [&](const float* W, __bf16* Wt, int Npad, int Nreal) {
    const long n = (long)Npad * 128;
    transpose_w_kernel<<<(int)((n + 255) / 256), 256, 0, stream>>>(W, Wt, Npad, Nreal);
  };
  for (int s = 0; s < 4; ++s) {
    tw(sets[s].wq,    pset[s].wqt,    CC, CC);
    tw(sets[s].wk,    pset[s].wkt,    CC, CC);
    tw(sets[s].wv,    pset[s].wvt,    CC, CC);
    tw(sets[s].wo,    pset[s].wot,    CC, CC);
    tw(sets[s].pe_w2, pset[s].pe_w2t, CC, CC);
    tw(sets[s].w_w1,  pset[s].w_w1t,  CC, CC);
    tw(sets[s].w_w2,  pset[s].w_w2t,  16, HH); // zero-padded columns 8..15
  }
  tw(wm, wmt, CC, CC);

  // ---- KNN ----
  knn_kernel<KQQ><<<NP_, 256, 0, stream>>>(t_coord, t_coord, NT_, NT_, knnqq);
  knn_kernel<KQC><<<NP_, 256, 0, stream>>>(t_coord, a_coord, NT_, NA_, knnqc);
  knn_kernel<KQC><<<NP_, 256, 0, stream>>>(a_coord, t_coord, NA_, NT_, knncq);

  // ---- initial features: fp32 copy + bf16 mirror ----
  hipMemcpyAsync(t0, t_feat, featB, hipMemcpyDeviceToDevice, stream);
  hipMemcpyAsync(a0, a_feat, featB, hipMemcpyDeviceToDevice, stream);
  {
    const long n = (long)NP_ * CC;
    cvt_bf16_kernel<<<(int)((n + 255) / 256), 256, 0, stream>>>(t_feat, t0b, n);
    cvt_bf16_kernel<<<(int)((n + 255) / 256), 256, 0, stream>>>(a_feat, a0b, n);
  }

  auto run_gemm = [&](const __bf16* A, const __bf16* Wt, const float* bias,
                      const float* resid, float* Y, __bf16* Ybf, int M, int N) {
    const int tiles = (M / 16) * (N / 16);
    gemm_kernel<<<(tiles + 7) / 8, 256, 0, stream>>>(A, Wt, bias, resid, Y, Ybf, M, N);
  };
  auto attn_call = [&](const PT& p, const PTB& pb,
                       const float* qfeat, const __bf16* qfeat_bf, const float* qcoord,
                       const __bf16* kvfeat_bf, const float* kvcoord, const int* knn,
                       int K, float* outfeat, __bf16* outfeat_bf) {
    run_gemm(qfeat_bf,  pb.wqt, nullptr, nullptr, qproj, nullptr, NP_, CC);
    run_gemm(kvfeat_bf, pb.wkt, nullptr, nullptr, kproj, nullptr, NP_, CC);
    run_gemm(kvfeat_bf, pb.wvt, nullptr, nullptr, vproj, nullptr, NP_, CC);
    if (K == KQQ)
      attn_kernel<KQQ><<<NP_, 256, 0, stream>>>(qproj, kproj, vproj, qcoord, kvcoord, knn,
          p.pe_w1, p.pe_b1, pb.pe_w2t, p.pe_b2, pb.w_w1t, p.w_b1, pb.w_w2t, p.w_b2, attout_bf);
    else
      attn_kernel<KQC><<<NP_, 256, 0, stream>>>(qproj, kproj, vproj, qcoord, kvcoord, knn,
          p.pe_w1, p.pe_b1, pb.pe_w2t, p.pe_b2, pb.w_w1t, p.w_b1, pb.w_w2t, p.w_b2, attout_bf);
    // residual: outfeat = qfeat + attout @ wo + bo   (writes fp32 + bf16 mirror)
    run_gemm(attout_bf, pb.wot, p.bo, qfeat, outfeat, outfeat_bf, NP_, CC);
  };

  float* tcur = t0; __bf16* tcurb = t0b; float* tnxt = t1; __bf16* tnxtb = t1b;
  float* acur = a0; __bf16* acurb = a0b; float* anxt = a1; __bf16* anxtb = a1b;
  for (int blk = 0; blk < 2; ++blk) {
    const PT& ctx = sets[2 * blk + 0]; const PTB& ctxb = pset[2 * blk + 0];
    const PT& qry = sets[2 * blk + 1]; const PTB& qryb = pset[2 * blk + 1];
    attn_call(qry, qryb, tcur, tcurb, t_coord, tcurb, t_coord, knnqq, KQQ, tnxt, tnxtb);
    { float* f = tcur; tcur = tnxt; tnxt = f; __bf16* h = tcurb; tcurb = tnxtb; tnxtb = h; }
    attn_call(ctx, ctxb, tcur, tcurb, t_coord, acurb, a_coord, knnqc, KQC, tnxt, tnxtb);
    { float* f = tcur; tcur = tnxt; tnxt = f; __bf16* h = tcurb; tcurb = tnxtb; tnxtb = h; }
    attn_call(ctx, ctxb, acur, acurb, a_coord, tcurb, t_coord, knncq, KQC, anxt, anxtb);
    { float* f = acur; acur = anxt; anxt = f; __bf16* h = acurb; acurb = anxtb; anxtb = h; }
  }

  // ---- dual-softmax matching ----
  run_gemm(tcurb, wmt, nullptr, nullptr, nullptr, fab, NP_, CC);
  run_gemm(acurb, wmt, nullptr, nullptr, nullptr, fbb, NP_, CC);
  const float scale = 1.0f / sqrtf((float)CC); // TEMP = 1.0
  {
    const long tiles = (long)BB * (NT_ / 16) * (NA_ / 16);
    sim_kernel<<<(int)((tiles + 7) / 8), 256, 0, stream>>>(fab, fbb, sim, scale);
  }
  row_stats_kernel<<<BB * NT_, 256, 0, stream>>>(sim, rmax, rsum);
  col_stats_kernel<<<(BB * NA_ + 255) / 256, 256, 0, stream>>>(sim, cmax, csum);
  {
    const long total = (long)BB * NT_ * NA_;
    conf_kernel<<<(int)((total + 255) / 256), 256, 0, stream>>>(
        sim, rmax, rsum, cmax, csum, (float*)d_out);
  }
}
